// conv_mamba_block_52596169506817
// MI455X (gfx1250) — matrix operand order repaired and checked by reading, not compiled, not run on hardware
//
#include <hip/hip_runtime.h>
#include <hip/hip_bf16.h>

// ---------------------------------------------------------------------------
// Problem constants (from reference): B=2, L=2048, D=512, DI=1024, N=16, K=4,
// R=32, CK=31, DFF=2048.  M = B*L = 4096 rows for every GEMM.
// ---------------------------------------------------------------------------
#define CB  2
#define CL  2048
#define CD  512
#define CDI 1024
#define CN  16
#define CKW 4
#define CR  32
#define CCK 31
#define CDFF 2048
#define CM  (CB * CL)   // 4096

// gfx1250 async global->LDS staging for the GEMM B tiles.
#define USE_ASYNC 1

typedef __attribute__((ext_vector_type(16))) _Float16 v16h;
typedef __attribute__((ext_vector_type(8)))  float    v8f;

__device__ __forceinline__ float sigmoidf_(float x) { return 1.f / (1.f + __expf(-x)); }
__device__ __forceinline__ float siluf_(float x)    { return x * sigmoidf_(x); }

// ---------------------------------------------------------------------------
// Weight convert + transpose: W (KxN f32 row-major) -> Wt (NxK f16 row-major)
// so B-matrix WMMA fragments become contiguous 16B loads per lane.
// ---------------------------------------------------------------------------
__global__ void wconvert_kernel(const float* __restrict__ src, _Float16* __restrict__ dst,
                                int K, int N) {
  size_t idx = (size_t)blockIdx.x * blockDim.x + threadIdx.x;
  if (idx >= (size_t)K * N) return;
  int k = (int)(idx / N);
  int n = (int)(idx % N);
  dst[(size_t)n * K + k] = (_Float16)src[idx];
}

// Strided f32 -> f16 (used to extract dt = dbl[..., :32])
__global__ void cvt_strided_kernel(const float* __restrict__ src, _Float16* __restrict__ dst,
                                   int rows, int cols, int srs) {
  size_t idx = (size_t)blockIdx.x * blockDim.x + threadIdx.x;
  if (idx >= (size_t)rows * cols) return;
  int r = (int)(idx / cols), c = (int)(idx % cols);
  dst[idx] = (_Float16)src[(size_t)r * srs + c];
}

// ---------------------------------------------------------------------------
// Cooperative staging of one 64col x 32k B tile (4 KB) into LDS in exact
// WMMA-fragment order. 256 threads x 16 B each. slot = tc*32 + lane; a lane's
// fragment for tile tc is then 32 contiguous bytes at (tc*32+lane)*32.
// Uses gfx1250 GLOBAL_LOAD_ASYNC_TO_LDS_B128 (ASYNCcnt) when USE_ASYNC.
// ---------------------------------------------------------------------------
__device__ __forceinline__ void stage_B(const _Float16* __restrict__ Bt, _Float16* sbuf,
                                        int col0, int K, int kb, int tid) {
  const int slot = tid >> 1, half = tid & 1;
  const int tc  = slot >> 5, l = slot & 31;
  const int col = col0 + tc * 16 + (l & 15);
  const int khi = l >> 4;
  const _Float16* g = Bt + (size_t)col * K + kb + 8 * khi + 16 * half;
  _Float16* dst = sbuf + slot * 16 + half * 8;
#if USE_ASYNC
  unsigned loff = (unsigned)(uintptr_t)dst;  // low 32 bits of generic ptr = LDS byte offset
  asm volatile("global_load_async_to_lds_b128 %0, %1, off" :: "v"(loff), "v"(g) : "memory");
#else
  *(uint4*)dst = *(const uint4*)g;
#endif
}

__device__ __forceinline__ void wait_async_barrier() {
#if USE_ASYNC
  asm volatile("s_wait_asynccnt 0x0" ::: "memory");
#endif
  __syncthreads();
}

// ---------------------------------------------------------------------------
// WMMA GEMM: out = A(MxK f16) * Bt^T(Bt is NxK f16) [+bias][epilogue][+resid]
// Wave tile: 16x64 (A fragment reused across 4 N-tiles). Block: 8 waves ->
// 128x64. B double-buffered through LDS (async DMA); A direct from global.
// Fragment layouts follow CDNA5 ISA 7.12.2 (wave32):
//   A 16x32 f16 : lane(r=lane&15, hi=lane>>4) holds K = hi*8 + [0..7] and
//                 16 + hi*8 + [0..7]  -> two contiguous 16B loads.
//   B 32x16 f16 : lane(col=lane&15, khi=lane>>4) holds K = khi*8 + [0..7] and
//                 16 + khi*8 + [0..7] (same order as A).
//   C/D 16x16 f32: VGPR j -> row = j + 8*(lane>>4), col = lane&15.
// EPI: 0=none, 1=bias, 2=bias+softplus, 3=bias+tanh-GELU
// ---------------------------------------------------------------------------
template <int EPI, bool OUT_HALF, bool RESID>
__global__ __launch_bounds__(256) void gemm_wmma_kernel(
    const _Float16* __restrict__ A, const _Float16* __restrict__ Bt,
    const float* __restrict__ bias, const float* __restrict__ resid,
    float* __restrict__ outF, _Float16* __restrict__ outH,
    int M, int N, int K) {
  __shared__ __align__(16) _Float16 sB[2][2048];  // 2 x 4KB double buffer

  const int lane = threadIdx.x & 31;
  const int wave = threadIdx.x >> 5;
  const int r    = lane & 15;
  const int hi   = lane >> 4;
  const int row0 = blockIdx.y * 128 + wave * 16;
  const int col0 = blockIdx.x * 64;

  v8f acc[4] = {};

  stage_B(Bt, sB[0], col0, K, 0, threadIdx.x);
  wait_async_barrier();

  const _Float16* arow = A + (size_t)(row0 + r) * K;
  for (int kb = 0; kb < K; kb += 32) {
    const int buf = (kb >> 5) & 1;
    if (kb + 32 < K) {
      stage_B(Bt, sB[buf ^ 1], col0, K, kb + 32, threadIdx.x);
      __builtin_prefetch(arow + kb + 32, 0, 1);  // global_prefetch_b8 (next A chunk)
    }
    union { uint4 q[2]; v16h h; } ua;
    ua.q[0] = *(const uint4*)(arow + kb + 8 * hi);
    ua.q[1] = *(const uint4*)(arow + kb + 16 + 8 * hi);
    union { uint4 q[2]; v16h h; } ub[4];
#pragma unroll
    for (int tc = 0; tc < 4; ++tc) {
      const uint4* ls = (const uint4*)(&sB[buf][(tc * 32 + lane) * 16]);
      ub[tc].q[0] = ls[0];
      ub[tc].q[1] = ls[1];
    }
#pragma unroll
    for (int tc = 0; tc < 4; ++tc)
      acc[tc] = __builtin_amdgcn_wmma_f32_16x16x32_f16(
          false, ua.h, false, ub[tc].h, (short)0, acc[tc], false, false);
    wait_async_barrier();  // close double-buffer window
  }

#pragma unroll
  for (int tc = 0; tc < 4; ++tc) {
    const int col = col0 + tc * 16 + r;
    float bv = 0.f;
    if constexpr (EPI >= 1) bv = bias[col];
#pragma unroll
    for (int j = 0; j < 8; ++j) {
      const int row = row0 + j + 8 * hi;
      float v = acc[tc][j] + bv;
      if constexpr (EPI == 2) v = (v > 20.f) ? v : log1pf(__expf(v));  // softplus
      if constexpr (EPI == 3) {                                        // tanh GELU
        float t = 0.7978845608028654f * (v + 0.044715f * v * v * v);
        v = 0.5f * v * (1.f + tanhf(t));
      }
      const size_t idx = (size_t)row * N + col;
      if constexpr (RESID) v += resid[idx];
      if constexpr (OUT_HALF) outH[idx] = (_Float16)v; else outF[idx] = v;
    }
  }
}

// ---------------------------------------------------------------------------
// LayerNorm over D=512 (one row per block, 256 threads, LDS tree reduction).
// Emits f16 (GEMM A operand) and optionally f32.
// ---------------------------------------------------------------------------
template <bool F32OUT>
__global__ __launch_bounds__(256) void ln512_kernel(
    const float* __restrict__ x, const float* __restrict__ g, const float* __restrict__ b,
    _Float16* __restrict__ o16, float* __restrict__ o32) {
  __shared__ float s1[256], s2[256];
  const int row = blockIdx.x, tid = threadIdx.x;
  const float* xr = x + (size_t)row * CD;
  const float v0 = xr[tid], v1 = xr[tid + 256];
  s1[tid] = v0 + v1;
  s2[tid] = v0 * v0 + v1 * v1;
  __syncthreads();
  for (int s = 128; s > 0; s >>= 1) {
    if (tid < s) { s1[tid] += s1[tid + s]; s2[tid] += s2[tid + s]; }
    __syncthreads();
  }
  const float mean = s1[0] * (1.f / CD);
  const float var  = s2[0] * (1.f / CD) - mean * mean;
  const float rstd = rsqrtf(var + 1e-5f);
  const float y0 = (v0 - mean) * rstd * g[tid] + b[tid];
  const float y1 = (v1 - mean) * rstd * g[tid + 256] + b[tid + 256];
  const size_t base = (size_t)row * CD;
  o16[base + tid] = (_Float16)y0;
  o16[base + tid + 256] = (_Float16)y1;
  if constexpr (F32OUT) { o32[base + tid] = y0; o32[base + tid + 256] = y1; }
}

// ---------------------------------------------------------------------------
// Causal depthwise conv K=4 + bias + SiLU. Input xi = xz[..., :DI]
// (row stride 2*DI). Writes xc f32 (scan needs it) and xc16 (GEMM A).
// ---------------------------------------------------------------------------
__global__ void dwconv4_silu_kernel(const float* __restrict__ xz, const float* __restrict__ w,
                                    const float* __restrict__ bias,
                                    float* __restrict__ xc, _Float16* __restrict__ xc16) {
  size_t idx = (size_t)blockIdx.x * blockDim.x + threadIdx.x;
  if (idx >= (size_t)CB * CL * CDI) return;
  const int d = (int)(idx % CDI);
  const int t = (int)((idx / CDI) % CL);
  const int b = (int)(idx / ((size_t)CDI * CL));
  float acc = bias[d];
#pragma unroll
  for (int k = 0; k < CKW; ++k) {
    const int tt = t - (CKW - 1) + k;
    if (tt >= 0) acc += w[d * CKW + k] * xz[((size_t)b * CL + tt) * (2 * CDI) + d];
  }
  const float v = siluf_(acc);
  xc[idx]   = v;
  xc16[idx] = (_Float16)v;
}

// Symmetric depthwise conv CK=31 + bias + SiLU -> f16 (GEMM A for cm_pw2).
__global__ void dwconv31_silu_kernel(const float* __restrict__ cin, const float* __restrict__ w,
                                     const float* __restrict__ bias, _Float16* __restrict__ o16) {
  size_t idx = (size_t)blockIdx.x * blockDim.x + threadIdx.x;
  if (idx >= (size_t)CB * CL * CD) return;
  const int d = (int)(idx % CD);
  const int t = (int)((idx / CD) % CL);
  const int b = (int)(idx / ((size_t)CD * CL));
  float acc = bias[d];
#pragma unroll
  for (int k = 0; k < CCK; ++k) {
    const int tt = t - (CCK - 1) / 2 + k;
    if (tt >= 0 && tt < CL) acc += w[d * CCK + k] * cin[((size_t)b * CL + tt) * CD + d];
  }
  o16[idx] = (_Float16)siluf_(acc);
}

// GLU: a_ * sigmoid(g_) from cpre (rows x 2D) -> cglu (rows x D) f32.
__global__ void glu_kernel(const float* __restrict__ cpre, float* __restrict__ cglu) {
  size_t idx = (size_t)blockIdx.x * blockDim.x + threadIdx.x;
  if (idx >= (size_t)CM * CD) return;
  const int row = (int)(idx / CD), c = (int)(idx % CD);
  const float a = cpre[(size_t)row * (2 * CD) + c];
  const float g = cpre[(size_t)row * (2 * CD) + CD + c];
  cglu[idx] = a * sigmoidf_(g);
}

// ---------------------------------------------------------------------------
// Selective scan, fused with D-skip, z-gating, and f16 emit for the W_out
// GEMM. Thread = one (b, d) channel; 16-entry SSM state lives in registers;
// Bm/Cm (shared across d) staged through LDS in 64-step chunks.
// ---------------------------------------------------------------------------
__global__ __launch_bounds__(256) void scan_kernel(
    const float* __restrict__ delta, const float* __restrict__ xc,
    const float* __restrict__ dbl, const float* __restrict__ A_log,
    const float* __restrict__ Dp, const float* __restrict__ xz,
    _Float16* __restrict__ y16) {
  __shared__ float sbc[64 * 32];
  const int b = blockIdx.y;
  const int d = blockIdx.x * 256 + threadIdx.x;

  float An[CN];
#pragma unroll
  for (int n = 0; n < CN; ++n) An[n] = -__expf(A_log[d * CN + n]);
  float hs[CN] = {};
  const float Dpv = Dp[d];
  const size_t base = (size_t)b * CL;

  for (int t0 = 0; t0 < CL; t0 += 64) {
    __syncthreads();
    for (int i = threadIdx.x; i < 64 * 32; i += 256) {
      const int tt = i >> 5, j = i & 31;  // j<16: Bm, j>=16: Cm
      sbc[i] = dbl[(base + t0 + tt) * (CR + 2 * CN) + CR + j];
    }
    __syncthreads();
    for (int tt = 0; tt < 64; ++tt) {
      const size_t rowi = base + t0 + tt;
      const float dv = delta[rowi * CDI + d];
      const float xv = xc[rowi * CDI + d];
      const float du = dv * xv;
      const float* sBv = &sbc[tt * 32];
      const float* sCv = sBv + CN;
      float y = 0.f;
#pragma unroll
      for (int n = 0; n < CN; ++n) {
        hs[n] = __expf(dv * An[n]) * hs[n] + du * sBv[n];
        y += hs[n] * sCv[n];
      }
      y += Dpv * xv;
      const float zv = xz[rowi * (2 * CDI) + CDI + d];
      y *= zv * (1.f / (1.f + __expf(-zv)));  // * silu(z)
      y16[rowi * CDI + d] = (_Float16)y;
    }
  }
}

// ---------------------------------------------------------------------------
// Host orchestration
// ---------------------------------------------------------------------------
extern "C" void kernel_launch(void* const* d_in, const int* in_sizes, int n_in,
                              void* d_out, int out_size, void* d_ws, size_t ws_size,
                              hipStream_t stream) {
  const float* x        = (const float*)d_in[0];
  const float* ln_g     = (const float*)d_in[1];
  const float* ln_b     = (const float*)d_in[2];
  const float* W_in     = (const float*)d_in[3];
  const float* conv_w   = (const float*)d_in[4];
  const float* conv_b   = (const float*)d_in[5];
  const float* W_x      = (const float*)d_in[6];
  const float* W_dt     = (const float*)d_in[7];
  const float* b_dt     = (const float*)d_in[8];
  const float* A_log    = (const float*)d_in[9];
  const float* Dp       = (const float*)d_in[10];
  const float* W_out    = (const float*)d_in[11];
  const float* cm_ln_g  = (const float*)d_in[12];
  const float* cm_ln_b  = (const float*)d_in[13];
  const float* cm_pw1_w = (const float*)d_in[14];
  const float* cm_pw1_b = (const float*)d_in[15];
  const float* cm_dw_w  = (const float*)d_in[16];
  const float* cm_dw_b  = (const float*)d_in[17];
  const float* cm_pw2_w = (const float*)d_in[18];
  const float* cm_pw2_b = (const float*)d_in[19];
  const float* ff_W1    = (const float*)d_in[20];
  const float* ff_b1    = (const float*)d_in[21];
  const float* ff_W2    = (const float*)d_in[22];
  const float* ff_b2    = (const float*)d_in[23];
  float* out = (float*)d_out;

  char* ws = (char*)d_ws;
  size_t cur = 0;
  auto alloc = [&](size_t bytes) -> void* {
    cur = (cur + 255) & ~(size_t)255;
    void* p = ws + cur;
    cur += bytes;
    return p;
  };

  // f16 transposed weights
  _Float16* WinT  = (_Float16*)alloc((size_t)CD * 2 * CDI * 2);
  _Float16* WxT   = (_Float16*)alloc((size_t)CDI * (CR + 2 * CN) * 2);
  _Float16* WdtT  = (_Float16*)alloc((size_t)CR * CDI * 2);
  _Float16* WoutT = (_Float16*)alloc((size_t)CDI * CD * 2);
  _Float16* Pw1T  = (_Float16*)alloc((size_t)CD * 2 * CD * 2);
  _Float16* Pw2T  = (_Float16*)alloc((size_t)CD * CD * 2);
  _Float16* FfW1T = (_Float16*)alloc((size_t)CD * CDFF * 2);
  _Float16* FfW2T = (_Float16*)alloc((size_t)CDFF * CD * 2);
  // f16 activations (GEMM A operands)
  _Float16* h16   = (_Float16*)alloc((size_t)CM * CD * 2);
  _Float16* xc16  = (_Float16*)alloc((size_t)CM * CDI * 2);
  _Float16* dt16  = (_Float16*)alloc((size_t)CM * CR * 2);
  _Float16* y16   = (_Float16*)alloc((size_t)CM * CDI * 2);
  _Float16* mln16 = (_Float16*)alloc((size_t)CM * CD * 2);
  _Float16* cc16  = (_Float16*)alloc((size_t)CM * CD * 2);
  _Float16* fln16 = (_Float16*)alloc((size_t)CM * CD * 2);
  _Float16* ffh16 = (_Float16*)alloc((size_t)CM * CDFF * 2);
  // f32 activations
  float* xz    = (float*)alloc((size_t)CM * 2 * CDI * 4);
  float* xc    = (float*)alloc((size_t)CM * CDI * 4);
  float* dbl   = (float*)alloc((size_t)CM * (CR + 2 * CN) * 4);
  float* delta = (float*)alloc((size_t)CM * CDI * 4);
  float* mbuf  = (float*)alloc((size_t)CM * CD * 4);
  float* cpre  = (float*)alloc((size_t)CM * 2 * CD * 4);
  float* cglu  = (float*)alloc((size_t)CM * CD * 4);
  float* m2    = (float*)alloc((size_t)CM * CD * 4);
  float* fbuf  = (float*)alloc((size_t)CM * CD * 4);
  (void)ws_size; (void)n_in; (void)in_sizes; (void)out_size;

  auto wcvt = [&](const float* src, _Float16* dst, int K, int N) {
    size_t n = (size_t)K * N;
    wconvert_kernel<<<(unsigned)((n + 255) / 256), 256, 0, stream>>>(src, dst, K, N);
  };
  wcvt(W_in,     WinT,  CD,   2 * CDI);
  wcvt(W_x,      WxT,   CDI,  CR + 2 * CN);
  wcvt(W_dt,     WdtT,  CR,   CDI);
  wcvt(W_out,    WoutT, CDI,  CD);
  wcvt(cm_pw1_w, Pw1T,  CD,   2 * CD);
  wcvt(cm_pw2_w, Pw2T,  CD,   CD);
  wcvt(ff_W1,    FfW1T, CD,   CDFF);
  wcvt(ff_W2,    FfW2T, CDFF, CD);

  // 1) h = LN(x) -> f16
  ln512_kernel<false><<<CM, 256, 0, stream>>>(x, ln_g, ln_b, h16, nullptr);
  // 2) xz = h @ W_in
  gemm_wmma_kernel<0, false, false><<<dim3((2 * CDI) / 64, CM / 128), 256, 0, stream>>>(
      h16, WinT, nullptr, nullptr, xz, nullptr, CM, 2 * CDI, CD);
  // 3) xc = silu(dwconv4(xi)) -> f32 + f16
  {
    size_t n = (size_t)CB * CL * CDI;
    dwconv4_silu_kernel<<<(unsigned)((n + 255) / 256), 256, 0, stream>>>(xz, conv_w, conv_b, xc, xc16);
  }
  // 4) dbl = xc @ W_x
  gemm_wmma_kernel<0, false, false><<<dim3((CR + 2 * CN) / 64, CM / 128), 256, 0, stream>>>(
      xc16, WxT, nullptr, nullptr, dbl, nullptr, CM, CR + 2 * CN, CDI);
  // 5) dt16 = f16(dbl[..., :R])
  {
    size_t n = (size_t)CM * CR;
    cvt_strided_kernel<<<(unsigned)((n + 255) / 256), 256, 0, stream>>>(dbl, dt16, CM, CR, CR + 2 * CN);
  }
  // 6) delta = softplus(dt @ W_dt + b_dt)
  gemm_wmma_kernel<2, false, false><<<dim3(CDI / 64, CM / 128), 256, 0, stream>>>(
      dt16, WdtT, b_dt, nullptr, delta, nullptr, CM, CDI, CR);
  // 7) selective scan + D-skip + z-gating -> y16
  scan_kernel<<<dim3(CDI / 256, CB), 256, 0, stream>>>(delta, xc, dbl, A_log, Dp, xz, y16);
  // 8) m = y @ W_out
  gemm_wmma_kernel<0, false, false><<<dim3(CD / 64, CM / 128), 256, 0, stream>>>(
      y16, WoutT, nullptr, nullptr, mbuf, nullptr, CM, CD, CDI);
  // 9) LN(m) -> f16
  ln512_kernel<false><<<CM, 256, 0, stream>>>(mbuf, cm_ln_g, cm_ln_b, mln16, nullptr);
  // 10) cpre = LN(m) @ cm_pw1 + b
  gemm_wmma_kernel<1, false, false><<<dim3((2 * CD) / 64, CM / 128), 256, 0, stream>>>(
      mln16, Pw1T, cm_pw1_b, nullptr, cpre, nullptr, CM, 2 * CD, CD);
  // 11) GLU
  {
    size_t n = (size_t)CM * CD;
    glu_kernel<<<(unsigned)((n + 255) / 256), 256, 0, stream>>>(cpre, cglu);
  }
  // 12) silu(dwconv31(cglu)) -> f16
  {
    size_t n = (size_t)CB * CL * CD;
    dwconv31_silu_kernel<<<(unsigned)((n + 255) / 256), 256, 0, stream>>>(cglu, cm_dw_w, cm_dw_b, cc16);
  }
  // 13) m2 = m + (cc @ cm_pw2 + b)   (residual fused)
  gemm_wmma_kernel<1, false, true><<<dim3(CD / 64, CM / 128), 256, 0, stream>>>(
      cc16, Pw2T, cm_pw2_b, mbuf, m2, nullptr, CM, CD, CD);
  // 14) f = LN(m2) -> f32 + f16
  ln512_kernel<true><<<CM, 256, 0, stream>>>(m2, ln_g, ln_b, fln16, fbuf);
  // 15) ffh = gelu(f @ ff_W1 + b1) -> f16 directly
  gemm_wmma_kernel<3, true, false><<<dim3(CDFF / 64, CM / 128), 256, 0, stream>>>(
      fln16, FfW1T, ff_b1, nullptr, nullptr, ffh16, CM, CDFF, CD);
  // 16) out = f + (ffh @ ff_W2 + b2)  (residual fused, writes d_out)
  gemm_wmma_kernel<1, false, true><<<dim3(CD / 64, CM / 128), 256, 0, stream>>>(
      ffh16, FfW2T, ff_b2, fbuf, out, nullptr, CM, CD, CDFF);
}
